// MultiHeadSelfAttention_10557029613731
// MI455X (gfx1250) — matrix operand order repaired and checked by reading, not compiled, not run on hardware
//
#include <hip/hip_runtime.h>
#include <hip/hip_bf16.h>

// ---------------------------------------------------------------------------
// MI455X (gfx1250) multi-head self-attention, f16 WMMA with f32 accumulation.
// Matrix math via V_WMMA_F32_16X16X32_F16; LDS tiles fed by the Tensor Data
// Mover (tensor_load_to_lds + s_wait_tensorcnt) with double buffering.
// ---------------------------------------------------------------------------

typedef _Float16 hf;
typedef __attribute__((ext_vector_type(16))) _Float16 v16h;
typedef __attribute__((ext_vector_type(8)))  float    v8f;
typedef unsigned int v4u __attribute__((ext_vector_type(4)));
typedef int          v8i __attribute__((ext_vector_type(8)));
typedef int          v4i __attribute__((ext_vector_type(4)));

#define WMMA_F16(a, b, c) \
    __builtin_amdgcn_wmma_f32_16x16x32_f16(false, (a), false, (b), (short)0, (c), false, false)

#if __has_builtin(__builtin_amdgcn_tensor_load_to_lds) && \
    __has_builtin(__builtin_amdgcn_s_wait_tensorcnt)
#define USE_TDM 1
#else
#define USE_TDM 0
#endif

#if USE_TDM
// Issue a 2-D TDM tile load (global -> LDS), D# per CDNA5 ISA ch.8.
//  tile0 x tile1 elements (2B each), source row stride = stride0 elements,
//  optional LDS row padding: pad DWORDs inserted at pad_interval boundaries.
__device__ __forceinline__ void tdm_load_tile(unsigned lds_byte_off, const hf* g,
                                              int tile0, int tile1,
                                              int tdim0, int tdim1,
                                              long long stride0,
                                              int pad_int_code, int pad_amt_code) {
    unsigned long long ga = (unsigned long long)g;
    v4u g0;
    g0[0] = 1u;                                              // count=1 (user D#)
    g0[1] = lds_byte_off;                                    // lds_addr
    g0[2] = (unsigned)(ga & 0xFFFFFFFFu);                    // global_addr[31:0]
    g0[3] = (unsigned)((ga >> 32) & 0x01FFFFFFu) | (2u << 30); // addr[56:32] | type=2
    v8i g1;
    g1[0] = (1 << 16)                                        // data_size = 2 bytes
          | (1 << 20)                                        // pad_enable
          | (pad_int_code << 22) | (pad_amt_code << 25);
    g1[1] = (tdim0 & 0xFFFF) << 16;                          // tensor_dim0[15:0]
    g1[2] = ((tdim0 >> 16) & 0xFFFF) | ((tdim1 & 0xFFFF) << 16);
    g1[3] = ((tdim1 >> 16) & 0xFFFF) | (tile0 << 16);        // tile_dim0
    g1[4] = (tile1 & 0xFFFF);                                // tile_dim1 (tile_dim2=0)
    g1[5] = (int)(stride0 & 0xFFFFFFFF);                     // tensor_dim0_stride lo
    g1[6] = (int)((stride0 >> 32) & 0xFFFF);                 // stride hi (dim1_stride=0)
    g1[7] = 0;
    v4i z4 = {0, 0, 0, 0};
#if defined(__clang_major__) && (__clang_major__ >= 23)
    v8i z8 = {0, 0, 0, 0, 0, 0, 0, 0};
    __builtin_amdgcn_tensor_load_to_lds(g0, g1, z4, z4, z8, 0);
#else
    __builtin_amdgcn_tensor_load_to_lds(g0, g1, z4, z4, 0);
#endif
}
#endif

// --- fragment loaders (layouts per CDNA5 ISA 7.12.2, wave32) ----------------
// A (16x32 MxK f16): lanes 0-15 -> M=lane, halves = {K 0..7, K 16..23};
//                    lanes 16-31 -> M=lane-16, halves = {K 8..15, K 24..31}.
__device__ __forceinline__ v16h load_afrag(const hf* base, int ldh) {
    const int lane = threadIdx.x & 31;
    const int m    = lane & 15;
    const int koff = (lane >> 4) << 3;            // 0 or 8
    const hf* p = base + (size_t)m * ldh + koff;
    union { v16h v; float4 f[2]; } u;
    u.f[0] = *(const float4*)(p);                 // K = koff .. koff+7
    u.f[1] = *(const float4*)(p + 16);            // K = 16+koff .. 16+koff+7
    return u.v;
}

// B (32x16 KxN f16): lanes 0-15 -> N=lane, halves = {K 0..7, K 16..23};
//                    lanes 16-31 -> N=lane-16, halves = {K 8..15, K 24..31}.
__device__ __forceinline__ v16h load_bfrag(const hf* base, int ldh) {
    const int lane = threadIdx.x & 31;
    const int nn = lane & 15;
    const int kb = (lane >> 4) << 3;              // 0 or 8
    const hf* p = base + (size_t)nn * ldh + kb;
    union { v16h v; float4 f[2]; } u;
    u.f[0] = *(const float4*)(p);
    u.f[1] = *(const float4*)(p + 16);
    return u.v;
}

// 16-lane-group reductions (rows of a C-tile live across one 16-lane half)
__device__ __forceinline__ float rmax16(float x) {
#pragma unroll
    for (int o = 8; o >= 1; o >>= 1) x = fmaxf(x, __shfl_xor(x, o, 32));
    return x;
}
__device__ __forceinline__ float rsum16(float x) {
#pragma unroll
    for (int o = 8; o >= 1; o >>= 1) x += __shfl_xor(x, o, 32);
    return x;
}

// --- fp32 -> f16 conversion -------------------------------------------------
__global__ void __launch_bounds__(256) cvt_f32_f16(const float* __restrict__ in,
                                                   hf* __restrict__ out, int n4) {
    int i = blockIdx.x * blockDim.x + threadIdx.x;
    if (i < n4) {
        float4 v = ((const float4*)in)[i];
        union { hf h[4]; float2 f; } u;
        u.h[0] = (hf)v.x; u.h[1] = (hf)v.y; u.h[2] = (hf)v.z; u.h[3] = (hf)v.w;
        ((float2*)out)[i] = u.f;
    }
}

// --- GEMM: C(MxN) = A(MxK,f16) * W(NxK,f16)^T, f32 accumulate ---------------
// 256 threads = 8 waves; block tile 128(M) x 64(N); wave = 16 rows x 4 N-tiles.
// B tiles double-buffered in LDS (row stride 72 halves, conflict-free); the
// TDM streams tile it+1 while the waves do WMMAs on tile it.
template <bool HALF_OUT>
__global__ void __launch_bounds__(256) gemm_wmma(const hf* __restrict__ A,
                                                 const hf* __restrict__ W,
                                                 void* __restrict__ Cout,
                                                 int M, int Nout, int K) {
    __shared__ __align__(16) hf sB[2][64 * 72];
    const int tid = threadIdx.x, wave = tid >> 5, lane = tid & 31;
    const int m0 = blockIdx.x * 128 + wave * 16;
    const int n0 = blockIdx.y * 64;
    const hf* wbase = W + (size_t)n0 * K;         // weight rows n0..n0+63
    const int NIT = K / 32;

    v8f c[4] = {};
#if USE_TDM
    // rows of 32 halves (16 DW) padded by 20 DW -> 72-half LDS stride
    if (wave == 0)
        tdm_load_tile((unsigned)(unsigned long long)&sB[0][0], wbase,
                      32, 64, K, Nout, K, /*pad 16DW interval*/3, /*20DW*/19);
    for (int it = 0; it < NIT; ++it) {
        __syncthreads();                          // buf[(it+1)&1] free to overwrite
        if (wave == 0) {
            if (it + 1 < NIT) {
                tdm_load_tile((unsigned)(unsigned long long)&sB[(it + 1) & 1][0],
                              wbase + (it + 1) * 32, 32, 64, K, Nout, K, 3, 19);
                __builtin_amdgcn_s_wait_tensorcnt(1);   // tile it done, it+1 in flight
            } else {
                __builtin_amdgcn_s_wait_tensorcnt(0);   // drain
            }
        }
        __syncthreads();                          // tile it visible to all waves
        const hf* bs = &sB[it & 1][0];
        if (it + 2 < NIT)                         // prefetch A two tiles ahead
            __builtin_prefetch(A + (size_t)(m0 + (lane & 15)) * K + (it + 2) * 32, 0, 0);
        v16h a = load_afrag(A + (size_t)m0 * K + it * 32, K);
#pragma unroll
        for (int t = 0; t < 4; t++) {
            v16h b = load_bfrag(bs + t * 16 * 72, 72);
            c[t] = WMMA_F16(a, b, c[t]);
        }
    }
#else
    for (int it = 0; it < NIT; ++it) {
        __syncthreads();
        {   // cooperative stage: 64 rows x 32 halves
            int row = tid >> 2, col = (tid & 3) << 3;
            float4 wv = *(const float4*)(wbase + (size_t)row * K + it * 32 + col);
            *(float4*)&sB[0][row * 72 + col] = wv;
        }
        __syncthreads();
        v16h a = load_afrag(A + (size_t)m0 * K + it * 32, K);
#pragma unroll
        for (int t = 0; t < 4; t++) {
            v16h b = load_bfrag(&sB[0][0] + t * 16 * 72, 72);
            c[t] = WMMA_F16(a, b, c[t]);
        }
    }
#endif
    // C layout: lane -> col (lane&15), reg r -> row r + 8*(lane>>4)
    const int nn = lane & 15, hi = lane >> 4;
#pragma unroll
    for (int t = 0; t < 4; t++)
#pragma unroll
        for (int r = 0; r < 8; r++) {
            size_t row = (size_t)m0 + r + 8 * hi;
            size_t col = (size_t)n0 + t * 16 + nn;
            if (HALF_OUT) ((hf*)Cout)[row * Nout + col] = (hf)c[t][r];
            else          ((float*)Cout)[row * Nout + col] = c[t][r];
        }
}

// --- Flash attention --------------------------------------------------------
// Grid: N*H*(L/128) blocks, 256 threads (8 waves). Each wave owns 16 query
// rows; block shares K/V 32-key tiles through LDS. Head dim 64, scale 1/8.
// K tile streamed by TDM (overlapped with the cooperative V-transpose stage).
__global__ void __launch_bounds__(256) attn_kernel(const hf* __restrict__ Q,
                                                   const hf* __restrict__ Kb,
                                                   const hf* __restrict__ Vb,
                                                   hf* __restrict__ O) {
    __shared__ __align__(16) hf sK [32 * 72];      // [key][d] row-major, padded
    __shared__ __align__(16) hf sVT[64 * 72];      // [d][key] (transposed)
    __shared__ __align__(16) hf sP [8 * 16 * 32];  // per-wave P tiles

    const int tid = threadIdx.x, wave = tid >> 5, lane = tid & 31;
    const int nn = lane & 15, hi = lane >> 4;
    const int b  = blockIdx.x;
    const int qc = b & 15;
    const int h  = (b >> 4) & 15;
    const int n  = b >> 8;
    const int hcol = h * 64;
    const size_t qrow0 = (size_t)n * 2048 + qc * 128 + wave * 16;

    const hf* qbase = Q + qrow0 * 1024 + hcol;
    v16h aq0 = load_afrag(qbase,      1024);       // d 0..31
    v16h aq1 = load_afrag(qbase + 32, 1024);       // d 32..63

    v8f acc[4] = {};
    float mstat[8], lstat[8];
#pragma unroll
    for (int r = 0; r < 8; r++) { mstat[r] = -INFINITY; lstat[r] = 0.f; }
    hf* pbase = sP + wave * (16 * 32);

    for (int jb = 0; jb < 2048; jb += 32) {
        __syncthreads();
        const size_t krow0 = (size_t)n * 2048 + jb;
#if USE_TDM
        // K tile: 32 rows x 64 halves (128B = 32 DW) padded by 4 DW -> 72-half stride
        if (wave == 0)
            tdm_load_tile((unsigned)(unsigned long long)&sK[0],
                          Kb + krow0 * 1024 + hcol, 64, 32, 1024, 8192, 1024,
                          /*interval 32DW*/4, /*pad 4DW*/3);
#endif
        {   // cooperative staging of V^T (and K when no TDM)
            int row = tid >> 3;                    // key 0..31
            int col = (tid & 7) << 3;              // d 0..56 step 8
            size_t krow = krow0 + row;
#if !USE_TDM
            float4 kv = *(const float4*)(Kb + krow * 1024 + hcol + col);
            *(float4*)&sK[row * 72 + col] = kv;
#endif
            float4 vv = *(const float4*)(Vb + krow * 1024 + hcol + col);
            union { float4 f; hf h8[8]; } u; u.f = vv;
#pragma unroll
            for (int i = 0; i < 8; i++) sVT[(col + i) * 72 + row] = u.h8[i];
        }
#if USE_TDM
        if (wave == 0) __builtin_amdgcn_s_wait_tensorcnt(0);
#endif
        __syncthreads();

        // S = Q K^T for 32 keys (two 16x16 tiles), K-dim 64 = 2 WMMAs each
        v8f s0 = {}, s1 = {};
        s0 = WMMA_F16(aq0, load_bfrag(sK,            72), s0);
        s0 = WMMA_F16(aq1, load_bfrag(sK + 32,       72), s0);
        s1 = WMMA_F16(aq0, load_bfrag(sK + 16 * 72,      72), s1);
        s1 = WMMA_F16(aq1, load_bfrag(sK + 16 * 72 + 32, 72), s1);

        // online softmax; reg r holds row (r + 8*hi), cols across 16 lanes
#pragma unroll
        for (int r = 0; r < 8; r++) {
            float x0 = s0[r] * 0.125f, x1 = s1[r] * 0.125f;
            float mx   = rmax16(fmaxf(x0, x1));
            float mnew = fmaxf(mstat[r], mx);
            float alpha = __expf(mstat[r] - mnew);
            float p0 = __expf(x0 - mnew), p1 = __expf(x1 - mnew);
            lstat[r] = lstat[r] * alpha + rsum16(p0 + p1);
            mstat[r] = mnew;
            acc[0][r] *= alpha; acc[1][r] *= alpha;
            acc[2][r] *= alpha; acc[3][r] *= alpha;
            int prow = r + 8 * hi;                 // C-layout -> LDS (row-major)
            pbase[prow * 32 + nn]      = (hf)p0;
            pbase[prow * 32 + 16 + nn] = (hf)p1;
        }
        asm volatile("s_wait_dscnt 0" ::: "memory"); // wave-local LDS RAW fence

        // O += P(16x32) * V(32x64): A-frag from staged P, B-frags from V^T
        v16h ap = load_afrag(pbase, 32);
#pragma unroll
        for (int t = 0; t < 4; t++) {
            v16h bv = load_bfrag(sVT + t * 16 * 72, 72);
            acc[t] = WMMA_F16(ap, bv, acc[t]);
        }
    }

#pragma unroll
    for (int t = 0; t < 4; t++)
#pragma unroll
        for (int r = 0; r < 8; r++) {
            size_t row = qrow0 + r + 8 * hi;
            O[row * 1024 + hcol + t * 16 + nn] = (hf)(acc[t][r] / lstat[r]);
        }
}

// ---------------------------------------------------------------------------
extern "C" void kernel_launch(void* const* d_in, const int* in_sizes, int n_in,
                              void* d_out, int out_size, void* d_ws, size_t ws_size,
                              hipStream_t stream) {
    const int M = 8192;     // N*L = 4*2048
    const int E = 1024;
    const size_t MB = 1024ull * 1024ull;

    char* ws = (char*)d_ws;
    // workspace layout (bytes): 7 f16 weights (14MB) | xh 16MB | t0 16MB | q,k,v 16MB each
    hf* wh[7];
    for (int i = 0; i < 7; i++) wh[i] = (hf*)(ws + (size_t)i * 2 * MB);
    hf* xh = (hf*)(ws + 14 * MB);
    hf* t0 = (hf*)(ws + 30 * MB);   // stage-1 tmp, later attention output
    hf* q  = (hf*)(ws + 46 * MB);
    hf* k  = (hf*)(ws + 62 * MB);
    hf* v  = (hf*)(ws + 78 * MB);

    // fp32 -> f16 conversions
    {
        int n4 = (M * E) / 4;
        cvt_f32_f16<<<n4 / 256, 256, 0, stream>>>((const float*)d_in[0], xh, n4);
    }
    for (int i = 0; i < 7; i++) {
        int n4 = (E * E) / 4;
        cvt_f32_f16<<<n4 / 256, 256, 0, stream>>>((const float*)d_in[1 + i], wh[i], n4);
    }

    dim3 gg(M / 128, E / 64);   // (64, 16)
    // q = (x Wq^T) Wqp^T ; k,v analogous
    gemm_wmma<true><<<gg, 256, 0, stream>>>(xh, wh[0], t0, M, E, E);
    gemm_wmma<true><<<gg, 256, 0, stream>>>(t0, wh[3], q,  M, E, E);
    gemm_wmma<true><<<gg, 256, 0, stream>>>(xh, wh[1], t0, M, E, E);
    gemm_wmma<true><<<gg, 256, 0, stream>>>(t0, wh[4], k,  M, E, E);
    gemm_wmma<true><<<gg, 256, 0, stream>>>(xh, wh[2], t0, M, E, E);
    gemm_wmma<true><<<gg, 256, 0, stream>>>(t0, wh[5], v,  M, E, E);

    // attention: 4 batches * 16 heads * 16 query-chunks = 1024 blocks
    attn_kernel<<<1024, 256, 0, stream>>>(q, k, v, t0);

    // out = att Wout^T  (fp32 straight to d_out)
    gemm_wmma<false><<<gg, 256, 0, stream>>>(t0, wh[6], d_out, M, E, E);
}
